// DecoderMultiHeadedAttention_58102317580621
// MI455X (gfx1250) — matrix operand order repaired and checked by reading, not compiled, not run on hardware
//
#include <hip/hip_runtime.h>

// ---------------- types ----------------
typedef __bf16 bf16_t;
typedef bf16_t v16bf __attribute__((ext_vector_type(16)));
typedef bf16_t v8bf  __attribute__((ext_vector_type(8)));
typedef float  v8f   __attribute__((ext_vector_type(8)));
typedef int    v4i_  __attribute__((ext_vector_type(4)));

#define B_   2
#define T_   2048
#define C_   128
#define H_   16
#define OUT_ 128
#define HC_  (C_*H_)   // 2048

#define NEG_INF (-__builtin_inff())

#if __has_builtin(__builtin_amdgcn_global_load_async_to_lds_b128) && \
    __has_builtin(__builtin_amdgcn_s_wait_asynccnt)
#define USE_ASYNC_LDS 1
#else
#define USE_ASYNC_LDS 0
#endif

__device__ __forceinline__ bf16_t f2bf(float f) {
  union { float f; unsigned u; } v; v.f = f;
  unsigned r = v.u + 0x7FFFu + ((v.u >> 16) & 1u);   // RNE
  union { unsigned short s; bf16_t h; } o; o.s = (unsigned short)(r >> 16);
  return o.h;
}

__device__ __forceinline__ v8f v8f_zero() {
  v8f z;
#pragma unroll
  for (int i = 0; i < 8; ++i) z[i] = 0.0f;
  return z;
}

// A-fragment (16x32 bf16, row-major source). lane_row already points at this
// lane's row (row = m = lane&15). Lanes hold 16 contiguous elements:
// k = hf*16..hf*16+15.
__device__ __forceinline__ v16bf load_a_frag(const bf16_t* lane_row, int k0, int hf) {
  const bf16_t* p = lane_row + k0 + hf * 16;
  v8bf lo = *(const v8bf*)p;
  v8bf hi = *(const v8bf*)(p + 8);
  return __builtin_shufflevector(lo, hi, 0,1,2,3,4,5,6,7,8,9,10,11,12,13,14,15);
}

// contiguous 16 bf16 (B-fragment half-lane chunk), 16B-aligned source
__device__ __forceinline__ v16bf load16(const bf16_t* p) {
  v8bf lo = *(const v8bf*)p;
  v8bf hi = *(const v8bf*)(p + 8);
  return __builtin_shufflevector(lo, hi, 0,1,2,3,4,5,6,7,8,9,10,11,12,13,14,15);
}

__device__ __forceinline__ v8f wmma_bf16(v16bf a, v16bf b, v8f c) {
  return __builtin_amdgcn_wmma_f32_16x16x32_bf16(false, a, false, b, (short)0, c, false, false);
}

__device__ __forceinline__ float redmax16(float v) {
  v = fmaxf(v, __shfl_xor(v, 1, 32));
  v = fmaxf(v, __shfl_xor(v, 2, 32));
  v = fmaxf(v, __shfl_xor(v, 4, 32));
  v = fmaxf(v, __shfl_xor(v, 8, 32));
  return v;
}
__device__ __forceinline__ float redsum16(float v) {
  v += __shfl_xor(v, 1, 32);
  v += __shfl_xor(v, 2, 32);
  v += __shfl_xor(v, 4, 32);
  v += __shfl_xor(v, 8, 32);
  return v;
}

// 16-byte global -> LDS copy (async on gfx1250 if builtin available)
__device__ __forceinline__ void cp16_g2l(const bf16_t* g, bf16_t* l) {
#if USE_ASYNC_LDS
  __builtin_amdgcn_global_load_async_to_lds_b128(
      (__attribute__((address_space(1))) v4i_*)(v4i_*)g,
      (__attribute__((address_space(3))) v4i_*)(v4i_*)l, 0, 0);
#else
  *(v8bf*)l = *(const v8bf*)g;
#endif
}

// ---------------- kernel 1: convert/pack ----------------
__global__ void pack_kernel(const float* __restrict__ x,
                            const float* __restrict__ Wq,
                            const float* __restrict__ Wk,
                            const float* __restrict__ Wv,
                            const float* __restrict__ Wfc,
                            bf16_t* __restrict__ xb,
                            bf16_t* __restrict__ wpkt,
                            bf16_t* __restrict__ wfct) {
  long i = (long)blockIdx.x * blockDim.x + threadIdx.x;
  const long NX = (long)B_ * T_ * C_;   // 524288
  const long NW = 3L * H_ * C_ * C_;    // 786432
  const long NF = (long)HC_ * OUT_;     // 262144
  if (i < NX) { xb[i] = f2bf(x[i]); return; }
  i -= NX;
  if (i < NW) {
    int cin  = (int)(i & (C_ - 1));
    int cout = (int)((i >> 7) & (C_ - 1));
    int h    = (int)((i >> 14) & (H_ - 1));
    int mat  = (int)(i >> 18);
    const float* W = (mat == 0) ? Wq : (mat == 1) ? Wk : Wv;
    wpkt[i] = f2bf(W[(long)cin * HC_ + cout * H_ + h]);
    return;
  }
  i -= NW;
  if (i < NF) {
    int cin  = (int)(i & (HC_ - 1));
    int cout = (int)(i >> 11);
    wfct[i] = f2bf(Wfc[(long)cin * OUT_ + cout]);
  }
}

// ---------------- kernel 2: QKV projection ----------------
__global__ void __launch_bounds__(128) qkv_kernel(const bf16_t* __restrict__ xb,
                                                  const bf16_t* __restrict__ wpkt,
                                                  bf16_t* __restrict__ q,
                                                  bf16_t* __restrict__ k,
                                                  bf16_t* __restrict__ vt) {
  int wid  = blockIdx.x * (blockDim.x >> 5) + (threadIdx.x >> 5);
  int lane = threadIdx.x & 31;
  int n  = lane & 15;
  int hf = lane >> 4;
  const int NT = T_ / 16;                     // 128
  const int TOT = 3 * B_ * H_ * NT;           // 12288
  if (wid >= TOT) return;

  int mat = wid / (B_ * H_ * NT);
  int rem = wid % (B_ * H_ * NT);
  int b   = rem / (H_ * NT);
  int h   = (rem / NT) % H_;
  int t0  = (rem % NT) * 16;

  const bf16_t* arow = xb + (size_t)(b * T_ + t0 + n) * C_;
  const bf16_t* wb   = wpkt + ((size_t)(mat * H_ + h)) * C_ * C_;

  v8f acc[8];
#pragma unroll
  for (int cb = 0; cb < 8; ++cb) acc[cb] = v8f_zero();

#pragma unroll
  for (int kk = 0; kk < 4; ++kk) {
    int k0 = kk * 32;
    v16bf a = load_a_frag(arow, k0, hf);
#pragma unroll
    for (int cb = 0; cb < 8; ++cb) {
      v16bf bf = load16(wb + (size_t)(cb * 16 + n) * C_ + k0 + hf * 16);
      acc[cb] = wmma_bf16(a, bf, acc[cb]);
    }
  }

  size_t head = (size_t)b * H_ + h;
  if (mat < 2) {
    bf16_t* dst = (mat == 0 ? q : k) + head * (size_t)T_ * C_;
#pragma unroll
    for (int cb = 0; cb < 8; ++cb)
#pragma unroll
      for (int r = 0; r < 8; ++r)
        dst[(size_t)(t0 + r + 8 * hf) * C_ + cb * 16 + n] = f2bf(acc[cb][r]);
  } else {
    bf16_t* dst = vt + head * (size_t)C_ * T_;
#pragma unroll
    for (int cb = 0; cb < 8; ++cb) {
      v8bf pk;
#pragma unroll
      for (int r = 0; r < 8; ++r) pk[r] = f2bf(acc[cb][r]);
      *(v8bf*)(dst + (size_t)(cb * 16 + n) * T_ + t0 + hf * 8) = pk;
    }
  }
}

// ---------------- kernel 3: flash attention ----------------
// Block = 4 waves = one (b,h) x 64 query rows; K/V panels (32 keys) staged
// once per block into LDS (async, double-buffered); per-wave 16x128 O tile.
#define KPAD 136   // 128 + 8 bf16 row padding (bank-conflict-free reads)
#define VPAD 40    // 32 + 8

__global__ void __launch_bounds__(128) attn_kernel(const bf16_t* __restrict__ q,
                                                   const bf16_t* __restrict__ k,
                                                   const bf16_t* __restrict__ vt,
                                                   bf16_t* __restrict__ att) {
  __shared__ __attribute__((aligned(64))) bf16_t kbuf[2][32][KPAD];
  __shared__ __attribute__((aligned(64))) bf16_t vbuf[2][128][VPAD];
  __shared__ __attribute__((aligned(64))) bf16_t plds[4][16][32];

  int tid  = threadIdx.x;
  int wv   = tid >> 5;
  int lane = tid & 31;
  int n  = lane & 15;
  int hf = lane >> 4;

  const int NBT = T_ / 64;                    // 32 blocks of 64 query rows
  int bid   = blockIdx.x;
  int b     = bid / (H_ * NBT);
  int h     = (bid / NBT) % H_;
  int t0blk = (bid % NBT) * 64;
  int t0    = t0blk + wv * 16;                // this wave's query tile

  size_t head = (size_t)b * H_ + h;
  const bf16_t* qh = q  + head * (size_t)T_ * C_;
  const bf16_t* kh = k  + head * (size_t)T_ * C_;
  const bf16_t* vh = vt + head * (size_t)C_ * T_;

  // Q tile as 4 A-fragments (contraction K = 0..127), loaded once.
  const bf16_t* arow = qh + (size_t)(t0 + n) * C_;
  v16bf aq[4];
#pragma unroll
  for (int kk = 0; kk < 4; ++kk) aq[kk] = load_a_frag(arow, kk * 32, hf);

  v8f oacc[8];
#pragma unroll
  for (int cb = 0; cb < 8; ++cb) oacc[cb] = v8f_zero();
  float rm[8], rl[8];
#pragma unroll
  for (int r = 0; r < 8; ++r) { rm[r] = NEG_INF; rl[r] = 0.0f; }

  const float scale = 0.08838834764831845f;   // 1/sqrt(128)
  int smax_w   = t0 + 15;                     // this wave's last visible key
  int smax_blk = t0blk + 63;                  // block's last visible key

  // ---- stage helper (cooperative, 128 threads, 8x16B per thread) ----
  auto stage = [&](int buf, int s0) {
#pragma unroll
    for (int j = 0; j < 4; ++j) {            // K: 32x128 (512 x 16B chunks)
      int ch = tid + 128 * j;
      int row = ch >> 4, col = ch & 15;
      cp16_g2l(kh + (size_t)(s0 + row) * C_ + col * 8,
               &kbuf[buf][row][col * 8]);
    }
#pragma unroll
    for (int j = 0; j < 4; ++j) {            // V: 128x32 (512 x 16B chunks)
      int ch = tid + 128 * j;
      int row = ch >> 2, col = ch & 3;
      cp16_g2l(vh + (size_t)row * T_ + s0 + col * 8,
               &vbuf[buf][row][col * 8]);
    }
  };

  stage(0, 0);                                // prologue

  int it = 0;
  for (int s0 = 0; s0 <= smax_blk; s0 += 32, ++it) {
    int cur = it & 1;
    bool has_next = (s0 + 32 <= smax_blk);
    if (has_next) stage(cur ^ 1, s0 + 32);
#if USE_ASYNC_LDS
    if (has_next) __builtin_amdgcn_s_wait_asynccnt(8);
    else          __builtin_amdgcn_s_wait_asynccnt(0);
#endif
    __syncthreads();                          // current panel visible to all

    if (s0 <= smax_w) {                       // wave-uniform
      // scores: two 16x16 tiles (keys s0..s0+15, s0+16..s0+31)
      v8f sA = v8f_zero(), sB = v8f_zero();
#pragma unroll
      for (int kk = 0; kk < 4; ++kk) {
        int k0 = kk * 32;
        v16bf bA = load16(&kbuf[cur][n][k0 + hf * 16]);
        sA = wmma_bf16(aq[kk], bA, sA);
        v16bf bB = load16(&kbuf[cur][16 + n][k0 + hf * 16]);
        sB = wmma_bf16(aq[kk], bB, sB);
      }

      bool need_mask = (s0 + 31 > t0);
      float pa[8], pb[8];
#pragma unroll
      for (int r = 0; r < 8; ++r) {
        float ta = sA[r], tb = sB[r];
        if (need_mask) {
          int t = t0 + r + 8 * hf;
          ta = (s0 + n      > t) ? NEG_INF : ta * scale;
          tb = (s0 + 16 + n > t) ? NEG_INF : tb * scale;
        } else { ta *= scale; tb *= scale; }
        pa[r] = ta; pb[r] = tb;
      }

      // online softmax; l kept per-lane (alpha uniform within half-wave)
#pragma unroll
      for (int r = 0; r < 8; ++r) {
        float mx = redmax16(fmaxf(pa[r], pb[r]));
        float nm = fmaxf(rm[r], mx);
        float al = __expf(rm[r] - nm);
        rm[r] = nm;
        pa[r] = __expf(pa[r] - nm);
        pb[r] = __expf(pb[r] - nm);
        rl[r] = rl[r] * al + pa[r] + pb[r];
#pragma unroll
        for (int cb = 0; cb < 8; ++cb) oacc[cb][r] *= al;
      }

      // D-layout -> A-layout for P via per-wave LDS
#pragma unroll
      for (int r = 0; r < 8; ++r) {
        int m = r + 8 * hf;
        plds[wv][m][n]      = f2bf(pa[r]);
        plds[wv][m][16 + n] = f2bf(pb[r]);
      }
      const bf16_t* pr = &plds[wv][lane & 15][0];
      v8bf lo = *(const v8bf*)(pr + hf * 16);
      v8bf hi = *(const v8bf*)(pr + hf * 16 + 8);
      v16bf ap = __builtin_shufflevector(lo, hi,
                   0,1,2,3,4,5,6,7,8,9,10,11,12,13,14,15);

      // O += P(16x32) @ V(32x128), V panel transposed in LDS
#pragma unroll
      for (int cb = 0; cb < 8; ++cb) {
        v16bf bv = load16(&vbuf[cur][cb * 16 + n][hf * 16]);
        oacc[cb] = wmma_bf16(ap, bv, oacc[cb]);
      }
    }

    __syncthreads();                          // safe to overwrite cur buffer
  }

  // normalize and store att[b, t, h*C + c] as bf16
#pragma unroll
  for (int r = 0; r < 8; ++r) {
    float inv = 1.0f / redsum16(rl[r]);
    int t = t0 + r + 8 * hf;
    size_t base = ((size_t)(b * T_ + t) * H_ + h) * C_;
#pragma unroll
    for (int cb = 0; cb < 8; ++cb)
      att[base + cb * 16 + n] = f2bf(oacc[cb][r] * inv);
  }
}

// ---------------- kernel 4: final FC ----------------
__global__ void __launch_bounds__(128) fc_kernel(const bf16_t* __restrict__ att,
                                                 const bf16_t* __restrict__ wfct,
                                                 const float* __restrict__ bfc,
                                                 float* __restrict__ out) {
  int wid  = blockIdx.x * (blockDim.x >> 5) + (threadIdx.x >> 5);
  int lane = threadIdx.x & 31;
  int n  = lane & 15;
  int hf = lane >> 4;
  const int TOT = (B_ * T_ / 16) * (OUT_ / 16);   // 2048
  if (wid >= TOT) return;

  int rt = wid >> 3;
  int cb = wid & 7;
  int row0 = rt * 16;
  int c0   = cb * 16;

  const bf16_t* arow = att  + (size_t)(row0 + n) * HC_;
  const bf16_t* wrow = wfct + (size_t)(c0 + n) * HC_;

  v8f acc = v8f_zero();
  for (int k0 = 0; k0 < HC_; k0 += 32) {
    v16bf a  = load_a_frag(arow, k0, hf);
    v16bf bf = load16(wrow + k0 + hf * 16);
    acc = wmma_bf16(a, bf, acc);
  }

  float bias = bfc[c0 + n];
#pragma unroll
  for (int r = 0; r < 8; ++r)
    out[(size_t)(row0 + r + 8 * hf) * OUT_ + c0 + n] = acc[r] + bias;
}

// ---------------- launcher ----------------
extern "C" void kernel_launch(void* const* d_in, const int* in_sizes, int n_in,
                              void* d_out, int out_size, void* d_ws, size_t ws_size,
                              hipStream_t stream) {
  (void)in_sizes; (void)n_in; (void)out_size; (void)ws_size;
  const float* x   = (const float*)d_in[0];
  const float* Wq  = (const float*)d_in[1];
  const float* Wk  = (const float*)d_in[2];
  const float* Wv  = (const float*)d_in[3];
  const float* Wfc = (const float*)d_in[4];
  const float* bfc = (const float*)d_in[5];
  float* out = (float*)d_out;

  char* ws = (char*)d_ws;
  size_t off = 0;
  bf16_t* xb   = (bf16_t*)(ws + off); off += (size_t)B_ * T_ * C_ * 2;        // 1 MB
  bf16_t* wpkt = (bf16_t*)(ws + off); off += (size_t)3 * H_ * C_ * C_ * 2;    // 1.5 MB
  bf16_t* wfct = (bf16_t*)(ws + off); off += (size_t)HC_ * OUT_ * 2;          // 0.5 MB
  bf16_t* qb   = (bf16_t*)(ws + off); off += (size_t)B_ * H_ * T_ * C_ * 2;   // 16 MB
  bf16_t* kb   = (bf16_t*)(ws + off); off += (size_t)B_ * H_ * T_ * C_ * 2;   // 16 MB
  bf16_t* vtb  = (bf16_t*)(ws + off); off += (size_t)B_ * H_ * C_ * T_ * 2;   // 16 MB
  bf16_t* attb = (bf16_t*)(ws + off); off += (size_t)B_ * T_ * H_ * C_ * 2;   // 16 MB

  pack_kernel<<<6144, 256, 0, stream>>>(x, Wq, Wk, Wv, Wfc, xb, wpkt, wfct);
  qkv_kernel<<<3072, 128, 0, stream>>>(xb, wpkt, qb, kb, vtb);
  attn_kernel<<<1024, 128, 0, stream>>>(qb, kb, vtb, attb);
  fc_kernel<<<512, 128, 0, stream>>>(attb, wfct, bfc, out);
}
